// EncoderBlock_19129784336423
// MI455X (gfx1250) — matrix operand order repaired and checked by reading, not compiled, not run on hardware
//
#include <hip/hip_runtime.h>
#include <math.h>

// ---------------------------------------------------------------------------
// Types
// ---------------------------------------------------------------------------
typedef __bf16 bf16;
typedef __attribute__((ext_vector_type(16))) __bf16        v16bf;
typedef __attribute__((ext_vector_type(8)))  float         v8f;
typedef __attribute__((ext_vector_type(4)))  unsigned int  u32x4;
typedef __attribute__((ext_vector_type(4)))  int           v4i;
typedef __attribute__((ext_vector_type(8)))  int           i32x8;
typedef __attribute__((ext_vector_type(4)))  int           i32x4;
typedef __attribute__((ext_vector_type(4)))  float         f32x4;

// Address-space-qualified int4 pointee types for the async-copy builtin
typedef v4i __attribute__((address_space(1))) gv4i;   // global
typedef v4i __attribute__((address_space(3))) lv4i;   // LDS

#define LDS_STRIDE 72   // 64 + 8 pad; rows are 144B (16B-aligned chunks, pad = 4 DWORDs)

// ---------------------------------------------------------------------------
// CDNA5 feature detection (probe-confirmed available on this toolchain)
// ---------------------------------------------------------------------------
#if defined(__HIP_DEVICE_COMPILE__) && defined(__has_builtin)
#if __has_builtin(__builtin_amdgcn_global_load_async_to_lds_b128) && \
    __has_builtin(__builtin_amdgcn_s_wait_asynccnt)
#define USE_ASYNC_COPY 1
#endif
#if __has_builtin(__builtin_amdgcn_tensor_load_to_lds) && \
    __has_builtin(__builtin_amdgcn_s_wait_tensorcnt)
#define USE_TDM 1
#endif
#endif
#ifndef USE_ASYNC_COPY
#define USE_ASYNC_COPY 0
#endif
#ifndef USE_TDM
#define USE_TDM 0
#endif

__device__ __forceinline__ void copy16_g2l(const void* g, void* l) {
#if USE_ASYNC_COPY
  __builtin_amdgcn_global_load_async_to_lds_b128((gv4i*)g, (lv4i*)l, 0, 0);
#else
  *(u32x4*)l = *(const u32x4*)g;
#endif
}
__device__ __forceinline__ void async_copy_wait() {
#if USE_ASYNC_COPY
  __builtin_amdgcn_s_wait_asynccnt(0);
#endif
}

#if USE_TDM
// Build a Tensor-DMA descriptor (D#) for a 2D bf16 tile [tile1 rows x tile0
// elems] of a row-major tensor [dim1 x dim0] (row stride stride0 elems) and
// issue TENSOR_LOAD_TO_LDS.  LDS destination gets 4 DWORDs of pad per row
// (pad_interval code 4 = 32 DWORDs, pad_amount code 3 = 4 DWORDs), matching
// LDS_STRIDE = 72 bf16 elements.  This toolchain's builtin takes 6 args:
// (g0 x4, g1 x8, g2 x4, g3 x4, extra x8, cpol) — groups 2/3 unused for 2D.
__device__ __forceinline__ void tdm_load_2d(void* lds, const void* gptr,
                                            unsigned dim0, unsigned dim1,
                                            unsigned long long stride0,
                                            unsigned tile0, unsigned tile1) {
  const unsigned long long ga = (unsigned long long)gptr;
  const unsigned lo =
      (unsigned)(unsigned long long)(__attribute__((address_space(3))) void*)lds;
  u32x4 g0;
  g0[0] = 1u;                                         // count=1, user mode
  g0[1] = lo;                                         // lds_addr (bytes)
  g0[2] = (unsigned)ga;                               // global_addr[31:0]
  g0[3] = (unsigned)((ga >> 32) & 0x01FFFFFFu) | (2u << 30);  // addr[56:32] | type=2
  i32x8 g1;
  g1[0] = (int)((1u << 16) | (1u << 20) | (4u << 22) | (3u << 25));
  //             data_size=2B  pad_en      interval=32DW  amount=4DW
  g1[1] = (int)((dim0 & 0xFFFFu) << 16);                              // dim0[15:0]
  g1[2] = (int)(((dim0 >> 16) & 0xFFFFu) | ((dim1 & 0xFFFFu) << 16)); // dim0[31:16], dim1[15:0]
  g1[3] = (int)(((dim1 >> 16) & 0xFFFFu) | (tile0 << 16));            // dim1[31:16], tile_dim0
  g1[4] = (int)tile1;                                                 // tile_dim1 (tile_dim2=0)
  g1[5] = (int)(unsigned)(stride0 & 0xFFFFFFFFull);                   // dim0_stride[31:0]
  g1[6] = (int)(unsigned)((stride0 >> 32) & 0xFFFFull);               // dim0_stride[47:32]
  g1[7] = 0;
  i32x4 z4 = {0, 0, 0, 0};
  i32x8 z8 = {0, 0, 0, 0, 0, 0, 0, 0};
  __builtin_amdgcn_tensor_load_to_lds(g0, g1, z4, z4, z8, 0);
}
#endif

// ---------------------------------------------------------------------------
// Transposing f32 -> bf16 weight conversion: out[N,K] = (bf16)in[K,N]^T
// ---------------------------------------------------------------------------
__global__ __launch_bounds__(256) void transpose_cvt_kernel(
    const float* __restrict__ in, bf16* __restrict__ out, int K, int N) {
  __shared__ float t[32][33];
  const int n0 = blockIdx.x * 32, k0 = blockIdx.y * 32;
  const int tx = threadIdx.x & 31, ty = threadIdx.x >> 5;  // 32 x 8
#pragma unroll
  for (int i = 0; i < 32; i += 8)
    t[ty + i][tx] = in[(size_t)(k0 + ty + i) * N + n0 + tx];
  __syncthreads();
#pragma unroll
  for (int i = 0; i < 32; i += 8)
    out[(size_t)(n0 + ty + i) * K + k0 + tx] = (bf16)t[tx][ty + i];
}

// ---------------------------------------------------------------------------
// LayerNorm over D=1024, one row per block (256 threads, 4 elems/thread),
// output bf16
// ---------------------------------------------------------------------------
__global__ __launch_bounds__(256) void layernorm_bf16_kernel(
    const float* __restrict__ x, const float* __restrict__ w,
    const float* __restrict__ b, bf16* __restrict__ out) {
  __shared__ float red[2][8];
  const int row = blockIdx.x;
  const float* xr = x + (size_t)row * 1024;
  f32x4 v = ((const f32x4*)xr)[threadIdx.x];
  float s  = v.x + v.y + v.z + v.w;
  float s2 = v.x * v.x + v.y * v.y + v.z * v.z + v.w * v.w;
#pragma unroll
  for (int off = 1; off < 32; off <<= 1) {
    s  += __shfl_xor(s,  off, 32);
    s2 += __shfl_xor(s2, off, 32);
  }
  const int wv = threadIdx.x >> 5, ln = threadIdx.x & 31;
  if (ln == 0) { red[0][wv] = s; red[1][wv] = s2; }
  __syncthreads();
  float ts = 0.f, ts2 = 0.f;
#pragma unroll
  for (int i = 0; i < 8; ++i) { ts += red[0][i]; ts2 += red[1][i]; }
  const float mean = ts * (1.0f / 1024.0f);
  const float var  = ts2 * (1.0f / 1024.0f) - mean * mean;
  const float rstd = rsqrtf(var + 1e-5f);
  const int c = threadIdx.x * 4;
  f32x4 wv4 = ((const f32x4*)w)[threadIdx.x];
  f32x4 bv4 = ((const f32x4*)b)[threadIdx.x];
  bf16* op = out + (size_t)row * 1024 + c;
  op[0] = (bf16)((v.x - mean) * rstd * wv4.x + bv4.x);
  op[1] = (bf16)((v.y - mean) * rstd * wv4.y + bv4.y);
  op[2] = (bf16)((v.z - mean) * rstd * wv4.z + bv4.z);
  op[3] = (bf16)((v.w - mean) * rstd * wv4.w + bv4.w);
}

// ---------------------------------------------------------------------------
// Tiled bf16 WMMA GEMM:  C[M,N] = A[M,K] * Wt[N,K]^T (+ bias) (+ epilogue)
//  - A row-major [M,K]; weights PRE-TRANSPOSED row-major [N,K] so both LDS
//    tiles are contiguous-row 2D tiles (TDM / async-b128 friendly).
//  - 64x64 output tile per 128-thread block (4 waves, each 16 rows x 64 cols)
//  EPI: 0 = bias -> bf16               (Q/K projections)
//       1 = bias, GELU -> bf16         (MLP up)
//       2 = bias + residual -> f32     (out-proj / MLP down)
//       3 = bias -> bf16, stored transposed out[col*M + row]  (V projection)
// ---------------------------------------------------------------------------
template <int EPI>
__global__ __launch_bounds__(128) void gemm_bf16_kernel(
    const bf16* __restrict__ A, const bf16* __restrict__ Wt,
    const float* __restrict__ bias, const float* __restrict__ res,
    bf16* __restrict__ outB, float* __restrict__ outF,
    int M, int N, int K) {
  __shared__ __align__(16) bf16 As[64 * LDS_STRIDE];  // [row][k]
  __shared__ __align__(16) bf16 Bt[64 * LDS_STRIDE];  // [n][k]

  const int tid   = threadIdx.x;
  const int lane  = tid & 31;
  const int wv    = tid >> 5;
  const int l15   = lane & 15;
  const int half  = lane >> 4;          // 0 for lanes 0-15, 1 for 16-31
  const int tileM = blockIdx.y * 64;
  const int tileN = blockIdx.x * 64;

  v8f acc[4] = {};

  for (int k0 = 0; k0 < K; k0 += 64) {
    __syncthreads();  // previous iteration done reading tiles
#if USE_TDM
    if (wv == 0) {
      tdm_load_2d(&As[0], &A[(size_t)tileM * K + k0], (unsigned)K, (unsigned)M,
                  (unsigned long long)K, 64u, 64u);
      tdm_load_2d(&Bt[0], &Wt[(size_t)tileN * K + k0], (unsigned)K, (unsigned)N,
                  (unsigned long long)K, 64u, 64u);
      __builtin_amdgcn_s_wait_tensorcnt(0);
    }
    __syncthreads();
#else
    for (int c = tid; c < 512; c += 128) {
      const int r = c >> 3, kc = (c & 7) << 3;
      copy16_g2l(&A[(size_t)(tileM + r) * K + k0 + kc], &As[r * LDS_STRIDE + kc]);
      copy16_g2l(&Wt[(size_t)(tileN + r) * K + k0 + kc], &Bt[r * LDS_STRIDE + kc]);
    }
    async_copy_wait();
    __syncthreads();
#endif

    if (k0 + 64 < K) {  // hint next tiles toward cache
      __builtin_prefetch(&A[(size_t)(tileM + (tid >> 1)) * K + k0 + 64 + (tid & 1) * 32], 0, 0);
      __builtin_prefetch(&Wt[(size_t)(tileN + (tid >> 1)) * K + k0 + 64 + (tid & 1) * 32], 0, 0);
    }

#pragma unroll
    for (int kk = 0; kk < 64; kk += 32) {
      // A fragment (16x32, MxK): lane row = l15, K split per half
      union { v16bf v; u32x4 u[2]; } af;
      {
        const int row = wv * 16 + l15;
        const int kf  = kk + (half << 4);
        af.u[0] = *(const u32x4*)&As[row * LDS_STRIDE + kf];       // K = kf..kf+7
        af.u[1] = *(const u32x4*)&As[row * LDS_STRIDE + kf + 8];   // K = kf+8..kf+15
      }
#pragma unroll
      for (int n = 0; n < 4; ++n) {
        // B fragment (32x16, KxN): lane col = l15, K = half*16 .. +15
        union { v16bf v; u32x4 u[2]; } bfr;
        const int col = n * 16 + l15;
        const int kf  = kk + (half << 4);
        bfr.u[0] = *(const u32x4*)&Bt[col * LDS_STRIDE + kf];
        bfr.u[1] = *(const u32x4*)&Bt[col * LDS_STRIDE + kf + 8];
        acc[n] = __builtin_amdgcn_wmma_f32_16x16x32_bf16(
            false, af.v, false, bfr.v, (short)0, acc[n], false, false);
      }
    }
  }

  // ---- epilogue (C layout: VGPR r -> row r + 8*half, lane l15 -> col) ----
#pragma unroll
  for (int n = 0; n < 4; ++n) {
    const int col = tileN + n * 16 + l15;
    const float bv = bias ? bias[col] : 0.0f;
#pragma unroll
    for (int r = 0; r < 8; ++r) {
      const int row = tileM + wv * 16 + r + (half << 3);
      float v = acc[n][r] + bv;
      if (EPI == 1) v = 0.5f * v * (1.0f + erff(v * 0.70710678118654752f));
      if (EPI == 2) {
        outF[(size_t)row * N + col] = v + res[(size_t)row * N + col];
      } else if (EPI == 3) {
        outB[(size_t)col * M + row] = (bf16)v;   // transposed store (V^T)
      } else {
        outB[(size_t)row * N + col] = (bf16)v;
      }
    }
  }
}

// ---------------------------------------------------------------------------
// Flash attention, bf16 WMMA, fp32 accumulate.
//  grid = (S/64, H, B); 128 threads = 4 waves; wave w owns 16 query rows.
//  Q/K layout: [B*S, H*Hd] bf16.  V is TRANSPOSED: vT[H*Hd, B*S] bf16, so PV
//  B-fragments are contiguous global loads (no LDS staging, no block
//  barriers — Ps is per-wave and per-wave DS ops are in order).
// ---------------------------------------------------------------------------
__global__ __launch_bounds__(128) void attention_kernel(
    const bf16* __restrict__ Q, const bf16* __restrict__ Km,
    const bf16* __restrict__ vT, bf16* __restrict__ O) {
  __shared__ __align__(16) bf16 Ps[4][16 * LDS_STRIDE];  // per-wave P [row][kv_local]

  const int tid  = threadIdx.x, lane = tid & 31, wv = tid >> 5;
  const int l15  = lane & 15, half = lane >> 4;
  const int qt   = blockIdx.x, h = blockIdx.y, b = blockIdx.z;
  const size_t rowBase = (size_t)b * 2048;
  const int colBase = h * 64;
  const int qRow0 = qt * 64 + wv * 16;

  union AV { v16bf v; u32x4 u[2]; };

  // Q fragments resident in registers for the whole pass (K = 0..31 / 32..63)
  AV aQ[2];
#pragma unroll
  for (int kk = 0; kk < 2; ++kk) {
    const int kf = kk * 32 + (half << 4);
    const bf16* qp = &Q[(rowBase + qRow0 + l15) * 1024 + colBase + kf];
    aQ[kk].u[0] = *(const u32x4*)qp;
    aQ[kk].u[1] = *(const u32x4*)(qp + 8);
  }

  v8f oAcc[4] = {};
  float mrow[8], lrow[8];
#pragma unroll
  for (int r = 0; r < 8; ++r) { mrow[r] = -1e30f; lrow[r] = 0.0f; }
  const float scale = 0.125f;  // 1/sqrt(64)

  for (int kt = 0; kt < 2048; kt += 64) {
    // ---- S = Q * K^T  (B fragment straight from global: K rows contiguous) ----
    v8f sc[4] = {};
#pragma unroll
    for (int n = 0; n < 4; ++n) {
#pragma unroll
      for (int kk = 0; kk < 2; ++kk) {
        AV bK;
        const bf16* kp =
            &Km[(rowBase + kt + n * 16 + l15) * 1024 + colBase + kk * 32 + (half << 4)];
        bK.u[0] = *(const u32x4*)kp;
        bK.u[1] = *(const u32x4*)(kp + 8);
        sc[n] = __builtin_amdgcn_wmma_f32_16x16x32_bf16(
            false, aQ[kk].v, false, bK.v, (short)0, sc[n], false, false);
      }
    }

    // ---- online softmax (row = r + 8*half; cols live across 16 lanes) ----
#pragma unroll
    for (int r = 0; r < 8; ++r) {
      float mx = fmaxf(fmaxf(sc[0][r], sc[1][r]), fmaxf(sc[2][r], sc[3][r]));
#pragma unroll
      for (int off = 1; off < 16; off <<= 1) mx = fmaxf(mx, __shfl_xor(mx, off, 32));
      const float mNew = fmaxf(mrow[r], mx * scale);
      const float corr = expf(mrow[r] - mNew);
      float psum = 0.0f;
#pragma unroll
      for (int n = 0; n < 4; ++n) {
        const float p = expf(sc[n][r] * scale - mNew);
        sc[n][r] = p;
        psum += p;
      }
#pragma unroll
      for (int off = 1; off < 16; off <<= 1) psum += __shfl_xor(psum, off, 32);
      lrow[r] = lrow[r] * corr + psum;
      mrow[r] = mNew;
#pragma unroll
      for (int n = 0; n < 4; ++n) oAcc[n][r] *= corr;
    }

    // ---- P: C-layout -> A-layout via per-wave LDS buffer (wave-local) ----
#pragma unroll
    for (int n = 0; n < 4; ++n)
#pragma unroll
      for (int r = 0; r < 8; ++r)
        Ps[wv][(r + (half << 3)) * LDS_STRIDE + n * 16 + l15] = (bf16)sc[n][r];

    // ---- O += P * V  (V^T rows contiguous in global) ----
#pragma unroll
    for (int kk = 0; kk < 2; ++kk) {
      AV aP;
      const int kf = kk * 32 + (half << 4);
      const bf16* pp = &Ps[wv][l15 * LDS_STRIDE + kf];
      aP.u[0] = *(const u32x4*)pp;
      aP.u[1] = *(const u32x4*)(pp + 8);
#pragma unroll
      for (int n = 0; n < 4; ++n) {
        AV bV;
        const bf16* vp =
            &vT[(size_t)(colBase + n * 16 + l15) * 4096 + rowBase + kt + kk * 32 + (half << 4)];
        bV.u[0] = *(const u32x4*)vp;
        bV.u[1] = *(const u32x4*)(vp + 8);
        oAcc[n] = __builtin_amdgcn_wmma_f32_16x16x32_bf16(
            false, aP.v, false, bV.v, (short)0, oAcc[n], false, false);
      }
    }
  }

  // ---- normalize + store O as bf16 in [B*S, H*Hd] ----
#pragma unroll
  for (int n = 0; n < 4; ++n)
#pragma unroll
    for (int r = 0; r < 8; ++r) {
      const int row = qRow0 + r + (half << 3);
      O[(rowBase + row) * 1024 + colBase + n * 16 + l15] = (bf16)(oAcc[n][r] / lrow[r]);
    }
}

// ---------------------------------------------------------------------------
// Host orchestration
// ---------------------------------------------------------------------------
extern "C" void kernel_launch(void* const* d_in, const int* in_sizes, int n_in,
                              void* d_out, int out_size, void* d_ws, size_t ws_size,
                              hipStream_t stream) {
  (void)in_sizes; (void)n_in; (void)out_size; (void)ws_size;
  const float* x    = (const float*)d_in[0];
  const float* ln1w = (const float*)d_in[1];
  const float* ln1b = (const float*)d_in[2];
  const float* wq   = (const float*)d_in[3];
  const float* bq   = (const float*)d_in[4];
  const float* wk   = (const float*)d_in[5];
  const float* bk   = (const float*)d_in[6];
  const float* wvp  = (const float*)d_in[7];
  const float* bv   = (const float*)d_in[8];
  const float* wo   = (const float*)d_in[9];
  const float* bo   = (const float*)d_in[10];
  const float* ln2w = (const float*)d_in[11];
  const float* ln2b = (const float*)d_in[12];
  const float* w1   = (const float*)d_in[13];
  const float* b1   = (const float*)d_in[14];
  const float* w2   = (const float*)d_in[15];
  const float* b2   = (const float*)d_in[16];
  float* out = (float*)d_out;

  const int BS = 4096, D = 1024, F = 4096;

  char* base = (char*)d_ws;
  size_t off = 0;
  auto carve = [&](size_t bytes) {
    void* r = base + off;
    off += (bytes + 255) & ~(size_t)255;
    return r;
  };
  bf16*  wq_t = (bf16*)carve((size_t)D * D * 2);   // [N,K] transposed weights
  bf16*  wk_t = (bf16*)carve((size_t)D * D * 2);
  bf16*  wv_t = (bf16*)carve((size_t)D * D * 2);
  bf16*  wo_t = (bf16*)carve((size_t)D * D * 2);
  bf16*  w1_t = (bf16*)carve((size_t)D * F * 2);
  bf16*  w2_t = (bf16*)carve((size_t)F * D * 2);
  bf16*  h_b  = (bf16*)carve((size_t)BS * D * 2);  // LN out (reused for LN2)
  bf16*  q_b  = (bf16*)carve((size_t)BS * D * 2);
  bf16*  k_b  = (bf16*)carve((size_t)BS * D * 2);
  bf16*  v_b  = (bf16*)carve((size_t)BS * D * 2);  // holds V^T [H*Hd, B*S]
  bf16*  o_b  = (bf16*)carve((size_t)BS * D * 2);
  float* x1   = (float*)carve((size_t)BS * D * 4);
  bf16*  g_b  = q_b;  // reuse q/k/v/o region (32MB) for MLP hidden [BS,F]

  // weights -> bf16, transposed to [N,K]
  transpose_cvt_kernel<<<dim3(D / 32, D / 32), 256, 0, stream>>>(wq, wq_t, D, D);
  transpose_cvt_kernel<<<dim3(D / 32, D / 32), 256, 0, stream>>>(wk, wk_t, D, D);
  transpose_cvt_kernel<<<dim3(D / 32, D / 32), 256, 0, stream>>>(wvp, wv_t, D, D);
  transpose_cvt_kernel<<<dim3(D / 32, D / 32), 256, 0, stream>>>(wo, wo_t, D, D);
  transpose_cvt_kernel<<<dim3(F / 32, D / 32), 256, 0, stream>>>(w1, w1_t, D, F);
  transpose_cvt_kernel<<<dim3(D / 32, F / 32), 256, 0, stream>>>(w2, w2_t, F, D);

  // pre-norm attention
  layernorm_bf16_kernel<<<BS, 256, 0, stream>>>(x, ln1w, ln1b, h_b);
  dim3 gD(D / 64, BS / 64);
  gemm_bf16_kernel<0><<<gD, 128, 0, stream>>>(h_b, wq_t, bq, nullptr, q_b, nullptr, BS, D, D);
  gemm_bf16_kernel<0><<<gD, 128, 0, stream>>>(h_b, wk_t, bk, nullptr, k_b, nullptr, BS, D, D);
  gemm_bf16_kernel<3><<<gD, 128, 0, stream>>>(h_b, wv_t, bv, nullptr, v_b, nullptr, BS, D, D);
  attention_kernel<<<dim3(32, 16, 2), 128, 0, stream>>>(q_b, k_b, v_b, o_b);
  gemm_bf16_kernel<2><<<gD, 128, 0, stream>>>(o_b, wo_t, bo, x, nullptr, x1, BS, D, D);

  // pre-norm MLP
  layernorm_bf16_kernel<<<BS, 256, 0, stream>>>(x1, ln2w, ln2b, h_b);
  gemm_bf16_kernel<1><<<dim3(F / 64, BS / 64), 128, 0, stream>>>(h_b, w1_t, b1, nullptr, g_b,
                                                                 nullptr, BS, F, D);
  gemm_bf16_kernel<2><<<gD, 128, 0, stream>>>(g_b, w2_t, b2, x1, nullptr, out, BS, D, F);
}
